// CardModel_GRUCell_34943853920892
// MI455X (gfx1250) — matrix operand order repaired and checked by reading, not compiled, not run on hardware
//
#include <hip/hip_runtime.h>

typedef __attribute__((ext_vector_type(16))) __bf16 v16bf;
typedef __attribute__((ext_vector_type(8)))  float  v8f;

namespace {
constexpr int kL = 512;
constexpr int kN = 4096;
constexpr int kI = 19;
constexpr int kH = 30;

constexpr int kGStride = 20;              // floats per gate-buffer column (16 rows + pad, conflict-free)
constexpr int kGSize   = 128 * kGStride;  // 128 logical columns
constexpr int kHStride = 34;              // floats per h row (30 cols + pad, conflict-free b64 frag loads)
constexpr int kHSize   = 16 * kHStride;
constexpr int kWaves   = 4;
}

// A-matrix (16x32 bf16) per-lane K index: VGPR p holds packed pair (j=0,1).
__device__ __forceinline__ int kA(int p, int hl, int j) {
  return (p < 4 ? 2 * p : 16 + 2 * (p - 4)) + 8 * hl + j;
}

// Fetch W^T(k, col) with gate-aligned column remap:
// cols [0,30)=r rows, [32,62)=z rows, [64,94)=n rows, rest zero-pad.
__device__ __forceinline__ float wfetch(const float* __restrict__ W, int Kdim, int k, int col) {
  int off = col & 31;
  int g   = col >> 5;
  if (off >= 30 || k >= Kdim) return 0.0f;
  return W[(g * 30 + off) * Kdim + k];
}

__device__ __forceinline__ float bfetch(const float* __restrict__ b, int col) {
  int off = col & 31;
  int g   = col >> 5;
  return (off >= 30) ? 0.0f : b[g * 30 + off];
}

__device__ __forceinline__ v8f splat8(float s) {
  v8f v;
#pragma unroll
  for (int i = 0; i < 8; ++i) v[i] = s;
  return v;
}

// Fast activations: v_exp_f32 + v_rcp_f32 (TRANS pipe), no IEEE divide sequences.
__device__ __forceinline__ float sigmoid_(float x) {
  return __builtin_amdgcn_rcpf(1.0f + __expf(-x));
}
__device__ __forceinline__ float tanh_(float x) {
  float e = __expf(2.0f * x);
  return (e - 1.0f) * __builtin_amdgcn_rcpf(e + 1.0f);
}

__device__ __forceinline__ v8f wmma_bf16(v16bf a, v16bf b, v8f c) {
  // 8 args: (neg_a, A, neg_b, B, c_mod, C, reuse_a, reuse_b)
  return __builtin_amdgcn_wmma_f32_16x16x32_bf16(false, a, false, b, (short)0, c, false, false);
}

// Branch-free x row fragment load (clamped address + select; K >= 19 forced to 0).
// Consecutive k within each half merge into global_load_b128 clauses.
__device__ __forceinline__ void load_xrow(const float* __restrict__ xr, int hl, float xv[16]) {
#pragma unroll
  for (int p = 0; p < 8; ++p) {
#pragma unroll
    for (int j = 0; j < 2; ++j) {
      const int k  = kA(p, hl, j);
      const int kk = (k < kI) ? k : 0;       // always in-bounds
      float v = xr[kk];
      xv[2 * p + j] = (k < kI) ? v : 0.0f;   // v_cndmask, no EXEC branch
    }
  }
}

__global__ __launch_bounds__(kWaves * 32) void gru_wmma_kernel(
    const float* __restrict__ x,  const float* __restrict__ h0,
    const float* __restrict__ Wih, const float* __restrict__ Whh,
    const float* __restrict__ bih, const float* __restrict__ bhh,
    const float* __restrict__ Wfc, const float* __restrict__ bfc,
    float* __restrict__ out)
{
  __shared__ __align__(16) float lds[kWaves * (kGSize + kHSize)];

  const int tid  = threadIdx.x;
  const int wave = tid >> 5;
  const int lane = tid & 31;
  const int n    = lane & 15;   // M / N index within 16-wide tile
  const int hl   = lane >> 4;   // lane half
  const int tile = blockIdx.x * kWaves + wave;   // 16-row batch tile id (0..255)
  const int row  = tile * 16 + n;                // this lane's global batch row

  float* Gw = lds + wave * (kGSize + kHSize);    // gate buffer, column-major [128][kGStride]
  float* Hw = Gw + kGSize;                       // hidden state f32 [16][kHStride]

  // ---- Load weight B-fragments once; they stay in VGPRs for the whole loop.
  // B (32x16) layout matches A: VGPR p holds K=kA(p,hl,0),kA(p,hl,0)+1 ; N=lane%16.
  v16bf Bih[6], Bhh[6];
#pragma unroll
  for (int c = 0; c < 6; ++c) {
    const int col = c * 16 + n;
    v16bf bi, bh;
#pragma unroll
    for (int p = 0; p < 8; ++p) {
      const int k0 = kA(p, hl, 0);
      bi[2 * p]     = (__bf16)wfetch(Wih, kI, k0,     col);
      bi[2 * p + 1] = (__bf16)wfetch(Wih, kI, k0 + 1, col);
      bh[2 * p]     = (__bf16)wfetch(Whh, kH, k0,     col);
      bh[2 * p + 1] = (__bf16)wfetch(Whh, kH, k0 + 1, col);
    }
    Bih[c] = bi;
    Bhh[c] = bh;
  }

  // ---- Per-lane bias values (depend only on column = c*16+n).
  float mb[4], bxn[2], bhn[2];
#pragma unroll
  for (int c = 0; c < 4; ++c) mb[c] = bfetch(bih, c * 16 + n) + bfetch(bhh, c * 16 + n);
#pragma unroll
  for (int c = 0; c < 2; ++c) {
    bxn[c] = bfetch(bih, (4 + c) * 16 + n);
    bhn[c] = bfetch(bhh, (4 + c) * 16 + n);
  }

  // ---- Initialize hidden state in LDS from h0 (f32).
#pragma unroll
  for (int e = 0; e < 15; ++e) {
    const int k = 15 * hl + e;
    Hw[n * kHStride + k] = h0[(size_t)row * kH + k];
  }

  // ---- Software pipeline: preload x for t=0; inside the loop issue loads for
  //      t+1 at the top so a full iteration (WMMAs + 480 activations) hides the
  //      global-memory latency.
  float xv[16];
  load_xrow(x + (size_t)row * kI, hl, xv);

  // ================== sequential time loop ==================
#pragma unroll 1
  for (int t = 0; t < kL; ++t) {
    // --- issue next step's x loads (clamped at the end; harmless reload).
    const int tn = (t + 1 < kL) ? (t + 1) : t;
    float xn[16];
    load_xrow(x + ((size_t)tn * kN + (size_t)row) * kI, hl, xn);

    // --- h A-fragment from LDS (f32 -> bf16); K pairs are 8B-aligned.
    v16bf Ah;
#pragma unroll
    for (int p = 0; p < 8; ++p) {
      const int k0  = kA(p, hl, 0);
      const int kk0 = (k0 < kH) ? k0 : 0;    // always in-bounds (even -> 8B aligned)
      float2 hp = *(const float2*)&Hw[n * kHStride + kk0];
      const float hx = (k0 < kH) ? hp.x : 0.0f;
      const float hy = (k0 < kH) ? hp.y : 0.0f;
      Ah[2 * p]     = (__bf16)hx;
      Ah[2 * p + 1] = (__bf16)hy;
    }

    // --- h-side WMMAs (start from bias accumulators).
    v8f acc0 = splat8(mb[0]), acc1 = splat8(mb[1]);
    v8f acc2 = splat8(mb[2]), acc3 = splat8(mb[3]);
    v8f ah0  = splat8(bhn[0]), ah1 = splat8(bhn[1]);
    acc0 = wmma_bf16(Ah, Bhh[0], acc0);
    acc1 = wmma_bf16(Ah, Bhh[1], acc1);
    acc2 = wmma_bf16(Ah, Bhh[2], acc2);
    acc3 = wmma_bf16(Ah, Bhh[3], acc3);
    ah0  = wmma_bf16(Ah, Bhh[4], ah0);
    ah1  = wmma_bf16(Ah, Bhh[5], ah1);

    // --- x-side WMMAs use xv loaded one iteration ago (latency fully hidden).
    v16bf Ax;
#pragma unroll
    for (int i = 0; i < 16; ++i) Ax[i] = (__bf16)xv[i];
    acc0 = wmma_bf16(Ax, Bih[0], acc0);
    acc1 = wmma_bf16(Ax, Bih[1], acc1);
    acc2 = wmma_bf16(Ax, Bih[2], acc2);
    acc3 = wmma_bf16(Ax, Bih[3], acc3);
    v8f ax0 = wmma_bf16(Ax, Bih[4], splat8(bxn[0]));
    v8f ax1 = wmma_bf16(Ax, Bih[5], splat8(bxn[1]));

    // --- Spill gate pre-activations to LDS (column-major; per lane the 8 rows are
    //     contiguous so these merge into ds_store_b128 pairs).
    // cols 0-31: r (merged), 32-63: z (merged), 64-95: i_n (+b_ih), 96-127: h_n (+b_hh).
#pragma unroll
    for (int v = 0; v < 8; v += 2) {
      const int rb = v + 8 * hl;  // row within tile
      *(float2*)&Gw[(0 * 16 + n) * kGStride + rb] = make_float2(acc0[v], acc0[v + 1]);
      *(float2*)&Gw[(1 * 16 + n) * kGStride + rb] = make_float2(acc1[v], acc1[v + 1]);
      *(float2*)&Gw[(2 * 16 + n) * kGStride + rb] = make_float2(acc2[v], acc2[v + 1]);
      *(float2*)&Gw[(3 * 16 + n) * kGStride + rb] = make_float2(acc3[v], acc3[v + 1]);
      *(float2*)&Gw[(4 * 16 + n) * kGStride + rb] = make_float2(ax0[v], ax0[v + 1]);
      *(float2*)&Gw[(5 * 16 + n) * kGStride + rb] = make_float2(ax1[v], ax1[v + 1]);
      *(float2*)&Gw[(6 * 16 + n) * kGStride + rb] = make_float2(ah0[v], ah0[v + 1]);
      *(float2*)&Gw[(7 * 16 + n) * kGStride + rb] = make_float2(ah1[v], ah1[v + 1]);
    }

    // --- Activations + h update: 480 elements / 32 lanes = 15 per lane.
    // Lane handles (row = n, k in [15*hl, 15*hl+15)). Single wave => LDS in-order, no barrier.
#pragma unroll
    for (int e = 0; e < 15; ++e) {
      const int k = 15 * hl + e;
      const float rp = Gw[(k)      * kGStride + n];
      const float zp = Gw[(32 + k) * kGStride + n];
      const float ip = Gw[(64 + k) * kGStride + n];
      const float hp = Gw[(96 + k) * kGStride + n];
      const float ho = Hw[n * kHStride + k];
      const float r  = sigmoid_(rp);
      const float z  = sigmoid_(zp);
      const float nn = tanh_(ip + r * hp);
      const float hn = nn + z * (ho - nn);   // (1-z)*n + z*h
      Hw[n * kHStride + k] = hn;
    }

    // --- rotate pipeline registers.
#pragma unroll
    for (int i = 0; i < 16; ++i) xv[i] = xn[i];
  }

  // ================== head: y = sigmoid(h @ Wfc^T + bfc) ==================
  {
    float acc = bfc[hl];
#pragma unroll
    for (int k = 0; k < kH; ++k) acc += Wfc[hl * kH + k] * Hw[n * kHStride + k];
    out[(size_t)row * 2 + hl] = sigmoid_(acc);
  }
  // h output (after the 8192 y floats)
#pragma unroll
  for (int e = 0; e < 15; ++e) {
    const int k = 15 * hl + e;
    out[(size_t)kN * 2 + (size_t)row * kH + k] = Hw[n * kHStride + k];
  }
}

extern "C" void kernel_launch(void* const* d_in, const int* in_sizes, int n_in,
                              void* d_out, int out_size, void* d_ws, size_t ws_size,
                              hipStream_t stream) {
  (void)in_sizes; (void)n_in; (void)out_size; (void)d_ws; (void)ws_size;
  const float* x   = (const float*)d_in[0];
  const float* h0  = (const float*)d_in[1];
  const float* Wih = (const float*)d_in[2];
  const float* Whh = (const float*)d_in[3];
  const float* bih = (const float*)d_in[4];
  const float* bhh = (const float*)d_in[5];
  const float* Wfc = (const float*)d_in[6];
  const float* bfc = (const float*)d_in[7];
  float* out = (float*)d_out;

  dim3 grid(kN / 16 / kWaves);   // 64 blocks
  dim3 block(kWaves * 32);       // 4 waves -> 4 independent 16-row tiles per block
  hipLaunchKernelGGL(gru_wmma_kernel, grid, block, 0, stream,
                     x, h0, Wih, Whh, bih, bhh, Wfc, bfc, out);
}
